// GroupedQueryAttention_40467181863298
// MI455X (gfx1250) — matrix operand order repaired and checked by reading, not compiled, not run on hardware
//
#include <hip/hip_runtime.h>

// ---------------------------------------------------------------------------
// GQA attention block for MI455X (gfx1250, wave32, WMMA, async global->LDS).
// All matmuls (QKV proj, Q.K^T, P.V, out proj) run on v_wmma_f32_16x16x32_bf16.
// ---------------------------------------------------------------------------

typedef __bf16 bf16;
typedef __attribute__((ext_vector_type(16))) bf16  v16bf;
typedef __attribute__((ext_vector_type(8)))  bf16  bf16x8;
typedef __attribute__((ext_vector_type(4)))  bf16  bf16x4;
typedef __attribute__((ext_vector_type(8)))  float v8f;
typedef __attribute__((ext_vector_type(4)))  float f32x4;
typedef __attribute__((ext_vector_type(4)))  int   v4i;

union FragAB { v16bf v; bf16x8 h[2]; };

constexpr int  Bc = 2, Sc = 2048, Dc = 2048, Hc = 16, KVHc = 4, HDc = 128;
constexpr int  Gc = Hc / KVHc;
constexpr float SCALE = 0.08838834764831845f;   // 1/sqrt(128)

#define AS1 __attribute__((address_space(1)))
#define AS3 __attribute__((address_space(3)))

#if __has_builtin(__builtin_amdgcn_global_load_async_to_lds_b128)
#define HAVE_ASYNC_LDS 1
#endif

// 16-byte global -> LDS copy: async (ASYNCcnt-tracked, no VGPR transit) when
// the toolchain exposes the gfx1250 builtin, else plain VGPR-mediated copy.
__device__ __forceinline__ void cp16_g2l(const bf16* g, bf16* l) {
#ifdef HAVE_ASYNC_LDS
    __builtin_amdgcn_global_load_async_to_lds_b128(
        (AS1 v4i*)g, (AS3 v4i*)l, 0, 0);
#else
    *(bf16x8*)l = *(const bf16x8*)g;
#endif
}
__device__ __forceinline__ void wait_async() {
#ifdef HAVE_ASYNC_LDS
#if __has_builtin(__builtin_amdgcn_s_wait_asynccnt)
    __builtin_amdgcn_s_wait_asynccnt(0);
#else
    asm volatile("s_wait_asynccnt 0x0" ::: "memory");
#endif
#endif
}

template <bool RA>
__device__ __forceinline__ v8f wmma_bb(const v16bf& a, const v16bf& b, const v8f& c) {
    return __builtin_amdgcn_wmma_f32_16x16x32_bf16(false, a, false, b, (short)0, c, RA, false);
}

__device__ __forceinline__ void storeC(float* p, float v) { *p = v; }
__device__ __forceinline__ void storeC(bf16*  p, float v) { *p = (bf16)v; }

// ---------------------------------------------------------------------------
// fp32 -> bf16 downconvert (one pass over x and weights so all GEMM staging
// becomes pure byte movement on the async path).
// ---------------------------------------------------------------------------
__global__ __launch_bounds__(256)
void cvt_bf16(const float* __restrict__ in, bf16* __restrict__ out, int n4)
{
    const int i = blockIdx.x * blockDim.x + threadIdx.x;
    if (i >= n4) return;
    f32x4 t = ((const f32x4*)in)[i];
    bf16x4 r = { (bf16)t[0], (bf16)t[1], (bf16)t[2], (bf16)t[3] };
    ((bf16x4*)out)[i] = r;
}

// ---------------------------------------------------------------------------
// GEMM: C[M,N] = A[M,K] @ W[N,K]^T  (bf16 A/W, fp32 accumulate)
// 256 threads = 8 waves, block tile 128x128, BK=32, double-buffered LDS with
// async global->LDS staging overlapped with the WMMA chain.
// ---------------------------------------------------------------------------
template <typename TC>
__global__ __launch_bounds__(256)
void gemm_bf16_wmma(const bf16* __restrict__ A, const bf16* __restrict__ W,
                    TC* __restrict__ C, int M, int N, int K)
{
    __shared__ bf16 As[2][128][32];
    __shared__ bf16 Bs[2][128][32];

    const int tid    = threadIdx.x;
    const int lane   = tid & 31;
    const int lane16 = lane & 15;
    const int hl     = lane >> 4;
    const int wv     = tid >> 5;
    const int wM     = wv & 3;     // 4 wave tiles along M (4*32 = 128)
    const int wN     = wv >> 2;    // 2 wave tiles along N (2*64 = 128)
    const int m0     = blockIdx.y * 128;
    const int n0     = blockIdx.x * 128;

    v8f acc[2][4];
#pragma unroll
    for (int i = 0; i < 2; ++i)
#pragma unroll
        for (int j = 0; j < 4; ++j)
            acc[i][j] = v8f{0.f,0.f,0.f,0.f,0.f,0.f,0.f,0.f};

    // stage one 128x32 tile of A and of W into buffer `buf` (512 x 16B each)
    auto stage = [&](int buf, int k0) {
        for (int c = tid; c < 512; c += 256) {          // 2 chunks per thread
            const int row = c >> 2;                     // 4 chunks of 8 per row
            const int col = (c & 3) << 3;
            cp16_g2l(&A[(long)(m0 + row) * K + k0 + col], &As[buf][row][col]);
            cp16_g2l(&W[(long)(n0 + row) * K + k0 + col], &Bs[buf][row][col]);
        }
    };

    stage(0, 0);
    int cur = 0;
    for (int k0 = 0; k0 < K; k0 += 32) {
        wait_async();          // my share of tile[cur] is in LDS
        __syncthreads();       // everyone's share is in LDS; prev reads done
        if (k0 + 32 < K) stage(cur ^ 1, k0 + 32);   // prefetch next tile

        FragAB a[2], b[4];
#pragma unroll
        for (int i = 0; i < 2; ++i) {
            const int mr = wM * 32 + i * 16 + lane16;
            a[i].h[0] = *(const bf16x8*)&As[cur][mr][hl * 8];
            a[i].h[1] = *(const bf16x8*)&As[cur][mr][16 + hl * 8];
        }
#pragma unroll
        for (int j = 0; j < 4; ++j) {
            const int nr = wN * 64 + j * 16 + lane16;
            b[j].h[0] = *(const bf16x8*)&Bs[cur][nr][hl * 8];
            b[j].h[1] = *(const bf16x8*)&Bs[cur][nr][16 + hl * 8];
        }
#pragma unroll
        for (int i = 0; i < 2; ++i) {   // reuse_a across the j-chain
            acc[i][0] = wmma_bb<false>(a[i].v, b[0].v, acc[i][0]);
            acc[i][1] = wmma_bb<true >(a[i].v, b[1].v, acc[i][1]);
            acc[i][2] = wmma_bb<true >(a[i].v, b[2].v, acc[i][2]);
            acc[i][3] = wmma_bb<true >(a[i].v, b[3].v, acc[i][3]);
        }
        cur ^= 1;
    }

#pragma unroll
    for (int i = 0; i < 2; ++i)
#pragma unroll
        for (int j = 0; j < 4; ++j)
#pragma unroll
            for (int r = 0; r < 8; ++r) {
                const long row = m0 + wM * 32 + i * 16 + r + 8 * hl;
                const long col = n0 + wN * 64 + j * 16 + lane16;
                storeC(&C[row * (long)N + col], acc[i][j][r]);
            }
}

// ---------------------------------------------------------------------------
// RoPE + fp32 -> bf16 (optionally folds the 1/sqrt(HD) score scale into Q).
// Layout: [(b*S+s), h*HD + d], row stride nheads*128.
// ---------------------------------------------------------------------------
__global__ __launch_bounds__(256)
void rope_to_bf16(const float* __restrict__ in, bf16* __restrict__ out,
                  int nheads, int seq, int total, float scale)
{
    const int idx = blockIdx.x * blockDim.x + threadIdx.x;
    if (idx >= total) return;
    const int d  = idx & (HDc - 1);
    const int s  = (idx / (HDc * nheads)) % seq;
    const int d2 = d & 63;
    const float inv = __expf(-0.14391156826902358f * (float)d2); // 10000^(-d2/64)
    const float ang = (float)s * inv;
    float sn, cs;
    __sincosf(ang, &sn, &cs);
    const float rot = (d < 64) ? -in[idx + 64] : in[idx - 64];
    out[idx] = (bf16)((in[idx] * cs + rot * sn) * scale);
}

// ---------------------------------------------------------------------------
// Flash attention (causal, GQA): grid (B*H, S/64), 128 threads (4 waves).
// Each wave owns 16 query rows; KV consumed in 32-row tiles.
// Q pre-scaled by 1/sqrt(HD) at RoPE time.
// ---------------------------------------------------------------------------
__global__ __launch_bounds__(128)
void gqa_flash_wmma(const bf16* __restrict__ Q, const bf16* __restrict__ Kb,
                    const bf16* __restrict__ Vb, bf16* __restrict__ O)
{
    __shared__ bf16 Qs[64][HDc];     // 16 KB
    __shared__ bf16 Ks[32][HDc];     //  8 KB
    __shared__ bf16 VTs[HDc][32];    //  8 KB (V transposed)
    __shared__ bf16 Ps[4][16][32];   //  4 KB per-wave P relayout patch

    const int tid    = threadIdx.x;
    const int lane   = tid & 31;
    const int lane16 = lane & 15;
    const int hl     = lane >> 4;
    const int w      = tid >> 5;

    const int bh  = blockIdx.x;
    const int b   = bh / Hc;
    const int h   = bh % Hc;
    const int kvh = h / Gc;
    const int q0  = blockIdx.y * 64;

    const long qstride = (long)Hc * HDc;    // 2048
    const long kstride = (long)KVHc * HDc;  // 512
    const bf16* qbase = Q  + ((long)b * Sc + q0) * qstride + (long)h * HDc;
    const bf16* kbase = Kb +  (long)b * Sc * kstride + (long)kvh * HDc;
    const bf16* vbase = Vb +  (long)b * Sc * kstride + (long)kvh * HDc;

    // ---- stage Q tile (64 x 128), async ----
    for (int c = tid; c < 1024; c += 128) {
        const int r  = c >> 4;
        const int co = (c & 15) * 8;
        cp16_g2l(&qbase[(long)r * qstride + co], &Qs[r][co]);
    }
    wait_async();
    __syncthreads();

    // ---- per-wave Q fragments, K dim = 4 chunks of 32 ----
    FragAB qa[4];
    {
        const int qr = w * 16 + lane16;
#pragma unroll
        for (int kc = 0; kc < 4; ++kc) {
            qa[kc].h[0] = *(const bf16x8*)&Qs[qr][kc * 32 + hl * 8];
            qa[kc].h[1] = *(const bf16x8*)&Qs[qr][kc * 32 + 16 + hl * 8];
        }
    }

    v8f o[8];
#pragma unroll
    for (int nc = 0; nc < 8; ++nc) o[nc] = v8f{0.f,0.f,0.f,0.f,0.f,0.f,0.f,0.f};
    float mrow[8], lrow[8];
#pragma unroll
    for (int r = 0; r < 8; ++r) { mrow[r] = -1e30f; lrow[r] = 0.f; }

    const int q0w = q0 + w * 16;

    for (int j0 = 0; j0 < q0 + 64; j0 += 32) {
        __syncthreads();
        // stage K (32x128, async) and V^T (128x32, VGPR transpose)
        for (int c = tid; c < 512; c += 128) {
            const int r  = c >> 4;
            const int co = (c & 15) * 8;
            cp16_g2l(&kbase[(long)(j0 + r) * kstride + co], &Ks[r][co]);
            bf16x8 vv = *(const bf16x8*)&vbase[(long)(j0 + r) * kstride + co];
#pragma unroll
            for (int i = 0; i < 8; ++i) VTs[co + i][r] = vv[i];
        }
        wait_async();
        __syncthreads();

        // ---- S = Q K^T for two 16-col subtiles; batch frag loads ----
        v8f s[2];
#pragma unroll
        for (int jt = 0; jt < 2; ++jt) {
            const int krow = jt * 16 + lane16;
            FragAB kf[4];
#pragma unroll
            for (int kc = 0; kc < 4; ++kc) {
                kf[kc].h[0] = *(const bf16x8*)&Ks[krow][kc * 32 + hl * 8];
                kf[kc].h[1] = *(const bf16x8*)&Ks[krow][kc * 32 + 16 + hl * 8];
            }
            v8f st = v8f{0.f,0.f,0.f,0.f,0.f,0.f,0.f,0.f};
            st = wmma_bb<false>(qa[0].v, kf[0].v, st);
            st = wmma_bb<false>(qa[1].v, kf[1].v, st);
            st = wmma_bb<false>(qa[2].v, kf[2].v, st);
            st = wmma_bb<false>(qa[3].v, kf[3].v, st);
            s[jt] = st;
        }

        // ---- online softmax (row = r + 8*hl, col = lane16) ----
#pragma unroll
        for (int r = 0; r < 8; ++r) {
            const int row = q0w + r + 8 * hl;
            const int c0  = j0 + lane16;
            const int c1  = j0 + 16 + lane16;
            const float x0 = (c0 > row) ? -1e30f : s[0][r];
            const float x1 = (c1 > row) ? -1e30f : s[1][r];
            float mt = fmaxf(x0, x1);
#pragma unroll
            for (int d = 1; d < 16; d <<= 1) mt = fmaxf(mt, __shfl_xor(mt, d, 16));
            const float mnew  = fmaxf(mrow[r], mt);
            const float alpha = __expf(mrow[r] - mnew);
            const float p0 = (c0 > row) ? 0.f : __expf(x0 - mnew);
            const float p1 = (c1 > row) ? 0.f : __expf(x1 - mnew);
            float ps = p0 + p1;
#pragma unroll
            for (int d = 1; d < 16; d <<= 1) ps += __shfl_xor(ps, d, 16);
            lrow[r] = lrow[r] * alpha + ps;
            mrow[r] = mnew;
#pragma unroll
            for (int nc = 0; nc < 8; ++nc) o[nc][r] *= alpha;
            Ps[w][r + 8 * hl][lane16]      = (bf16)p0;
            Ps[w][r + 8 * hl][16 + lane16] = (bf16)p1;
        }

        // ---- O += P (16x32) @ V (32x128); same A -> reuse_a hint ----
        FragAB pa;
        pa.h[0] = *(const bf16x8*)&Ps[w][lane16][hl * 8];
        pa.h[1] = *(const bf16x8*)&Ps[w][lane16][16 + hl * 8];
#pragma unroll
        for (int g = 0; g < 2; ++g) {
            FragAB vf[4];
#pragma unroll
            for (int q = 0; q < 4; ++q) {
                const int nc = g * 4 + q;
                vf[q].h[0] = *(const bf16x8*)&VTs[nc * 16 + lane16][hl * 8];
                vf[q].h[1] = *(const bf16x8*)&VTs[nc * 16 + lane16][16 + hl * 8];
            }
            const int n0g = g * 4;
            if (g == 0) {
                o[0] = wmma_bb<false>(pa.v, vf[0].v, o[0]);
                o[1] = wmma_bb<true >(pa.v, vf[1].v, o[1]);
                o[2] = wmma_bb<true >(pa.v, vf[2].v, o[2]);
                o[3] = wmma_bb<true >(pa.v, vf[3].v, o[3]);
            } else {
                o[n0g + 0] = wmma_bb<true>(pa.v, vf[0].v, o[n0g + 0]);
                o[n0g + 1] = wmma_bb<true>(pa.v, vf[1].v, o[n0g + 1]);
                o[n0g + 2] = wmma_bb<true>(pa.v, vf[2].v, o[n0g + 2]);
                o[n0g + 3] = wmma_bb<true>(pa.v, vf[3].v, o[n0g + 3]);
            }
        }
    }

    // ---- epilogue: O / l, write bf16 [B*S, H*HD] ----
    bf16* obase = O + ((long)b * Sc + q0w) * qstride + (long)h * HDc;
#pragma unroll
    for (int nc = 0; nc < 8; ++nc)
#pragma unroll
        for (int r = 0; r < 8; ++r) {
            const int rr = r + 8 * hl;
            obase[(long)rr * qstride + nc * 16 + lane16] = (bf16)(o[nc][r] / lrow[r]);
        }
}

// ---------------------------------------------------------------------------
// Host-side orchestration.  Workspace layout (116 MB):
//   [  0, 16M) x bf16        [ 16, 24M) Wq bf16   [ 24, 26M) Wk bf16
//   [ 26, 28M) Wv bf16       [ 28, 36M) Wo bf16
//   [ 36, 68M) q fp32        [ 68, 76M) k fp32
//   [ 76, 92M) q bf16 (RoPE) [ 92, 96M) k bf16    [ 96,100M) v bf16
//   [100,116M) attn out bf16
// ---------------------------------------------------------------------------
extern "C" void kernel_launch(void* const* d_in, const int* in_sizes, int n_in,
                              void* d_out, int out_size, void* d_ws, size_t ws_size,
                              hipStream_t stream)
{
    const float* x  = (const float*)d_in[0];
    const float* Wq = (const float*)d_in[1];
    const float* Wk = (const float*)d_in[2];
    const float* Wv = (const float*)d_in[3];
    const float* Wo = (const float*)d_in[4];
    float* out = (float*)d_out;

    char* ws = (char*)d_ws;
    bf16*  xb  = (bf16*) (ws);
    bf16*  Wqb = (bf16*) (ws + (16ull  << 20));
    bf16*  Wkb = (bf16*) (ws + (24ull  << 20));
    bf16*  Wvb = (bf16*) (ws + (26ull  << 20));
    bf16*  Wob = (bf16*) (ws + (28ull  << 20));
    float* qf  = (float*)(ws + (36ull  << 20));
    float* kf  = (float*)(ws + (68ull  << 20));
    bf16*  qb  = (bf16*) (ws + (76ull  << 20));
    bf16*  kb  = (bf16*) (ws + (92ull  << 20));
    bf16*  vb  = (bf16*) (ws + (96ull  << 20));
    bf16*  ab  = (bf16*) (ws + (100ull << 20));

    const int M = Bc * Sc;          // 4096
    const dim3 blk(256);

    // downconvert activations + weights once
    cvt_bf16<<<(M * Dc / 4) / 256, 256, 0, stream>>>(x,  xb,  M * Dc / 4);
    cvt_bf16<<<(Hc * HDc * Dc / 4) / 256, 256, 0, stream>>>(Wq, Wqb, Hc * HDc * Dc / 4);
    cvt_bf16<<<(KVHc * HDc * Dc / 4) / 256, 256, 0, stream>>>(Wk, Wkb, KVHc * HDc * Dc / 4);
    cvt_bf16<<<(KVHc * HDc * Dc / 4) / 256, 256, 0, stream>>>(Wv, Wvb, KVHc * HDc * Dc / 4);
    cvt_bf16<<<(Dc * Hc * HDc / 4) / 256, 256, 0, stream>>>(Wo, Wob, Dc * Hc * HDc / 4);

    // projections: y = x @ W^T
    gemm_bf16_wmma<float><<<dim3((Hc * HDc) / 128, M / 128), blk, 0, stream>>>(
        xb, Wqb, qf, M, Hc * HDc, Dc);
    gemm_bf16_wmma<float><<<dim3((KVHc * HDc) / 128, M / 128), blk, 0, stream>>>(
        xb, Wkb, kf, M, KVHc * HDc, Dc);
    gemm_bf16_wmma<bf16><<<dim3((KVHc * HDc) / 128, M / 128), blk, 0, stream>>>(
        xb, Wvb, vb, M, KVHc * HDc, Dc);

    // RoPE + downconvert (score scale folded into Q)
    const int qtot = M * Hc * HDc;
    rope_to_bf16<<<qtot / 256, 256, 0, stream>>>(qf, qb, Hc, Sc, qtot, SCALE);
    const int ktot = M * KVHc * HDc;
    rope_to_bf16<<<ktot / 256, 256, 0, stream>>>(kf, kb, KVHc, Sc, ktot, 1.0f);

    // causal GQA flash attention
    gqa_flash_wmma<<<dim3(Bc * Hc, Sc / 64), dim3(128), 0, stream>>>(qb, kb, vb, ab);

    // output projection -> fp32 result
    gemm_bf16_wmma<float><<<dim3(Dc / 128, M / 128), blk, 0, stream>>>(
        ab, Wob, out, M, Dc, Hc * HDc);
}
